// CLFusion_86337432584664
// MI455X (gfx1250) — matrix operand order repaired and checked by reading, not compiled, not run on hardware
//
#include <hip/hip_runtime.h>

// ---------------------------------------------------------------- constants
#define N_ST   8192
#define DD     512      // ST_DIM
#define N_MM   8192
#define MMD    768      // MM_DIM
#define TOPKN  6
#define BN_EPS 1e-5f
// TAU = 0.5  ->  exp(s/TAU) == exp(2*s)

// ---------------------------------------------------------------- WMMA types
typedef _Float16 v16h __attribute__((ext_vector_type(16)));
typedef float    v8f  __attribute__((ext_vector_type(8)));
typedef unsigned int u32x4 __attribute__((ext_vector_type(4)));
typedef int i32x8 __attribute__((ext_vector_type(8)));
typedef int i32x4 __attribute__((ext_vector_type(4)));

union HFrag { v16h v; uint4 q[2]; };

// A fragment (16x32 f16, MxK): lane m = lane&15, half = lane>>4.
// elements 0..7  -> K = half*8 + 0..7 ; elements 8..15 -> K = 16 + half*8 + 0..7
__device__ __forceinline__ v16h load_a_frag(const _Float16* row_ptr, int k0, int half) {
    HFrag f;
    const _Float16* p = row_ptr + k0 + half * 8;
    f.q[0] = *(const uint4*)(p);
    f.q[1] = *(const uint4*)(p + 16);
    return f.v;
}

// B fragment (32x16 f16, KxN) from row-major Bt (Bt[n][k] = B[k][n]):
// lane n = lane&15, kg = lane>>4; elements 0..7 -> K = k0 + kg*8 + 0..7 ; elements 8..15 -> K = k0 + 16 + kg*8 + 0..7
__device__ __forceinline__ v16h load_b_frag(const _Float16* col_ptr, int k0, int kg) {
    HFrag f;
    const _Float16* p = col_ptr + k0 + kg * 8;
    f.q[0] = *(const uint4*)(p);
    f.q[1] = *(const uint4*)(p + 16);
    return f.v;
}

__device__ __forceinline__ v8f wmma_f16(v16h a, v16h b, v8f c) {
    return __builtin_amdgcn_wmma_f32_16x16x32_f16(false, a, false, b, (short)0, c,
                                                  false, false);
}

__device__ __forceinline__ v8f gemm_tile(const _Float16* A, int arow, int lda,
                                         const _Float16* Bt, int brow, int ldb,
                                         int K, int half) {
    const _Float16* ap = A + (long)arow * lda;
    const _Float16* bp = Bt + (long)brow * ldb;
    v8f c = {};
    for (int k0 = 0; k0 < K; k0 += 32)
        c = wmma_f16(load_a_frag(ap, k0, half), load_b_frag(bp, k0, half), c);
    return c;
}

// ---------------------------------------------------------------- TDM (Tensor Data Mover)
// 2-D f16 tile load: tile_d0 contiguous elements (x) by tile_d1 rows (y),
// row pitch = stride0 elements, packed row-major into LDS at lds_off.
// D# per CDNA5 ISA ch.8: group0 = {count|flags, lds_addr, global_addr, type=2},
// group1 = {mask/data_size, tensor dims, tile dims, dim0 stride}.
__device__ __forceinline__ void tdm_load_2d(unsigned lds_off, const void* gptr,
                                            unsigned tensor_d0, unsigned tensor_d1,
                                            unsigned tile_d0, unsigned tile_d1,
                                            unsigned stride0) {
    unsigned long long ga = (unsigned long long)gptr;
    u32x4 g0;
    g0[0] = 1u;                                              // count=1, user desc
    g0[1] = lds_off;                                         // LDS byte address
    g0[2] = (unsigned)ga;                                    // global_addr[31:0]
    g0[3] = (unsigned)((ga >> 32) & 0x01FFFFFFu) | (2u << 30); // addr[56:32] | type=2
    i32x8 g1;
    g1[0] = (int)(1u << 16);                                 // data_size=1 (2B), mask=0
    g1[1] = (int)((tensor_d0 & 0xFFFFu) << 16);              // barrier_addr=0 | dim0[15:0]
    g1[2] = (int)((tensor_d0 >> 16) | ((tensor_d1 & 0xFFFFu) << 16));
    g1[3] = (int)((tensor_d1 >> 16) | (tile_d0 << 16));      // dim1[31:16] | tile_dim0
    g1[4] = (int)tile_d1;                                    // tile_dim1 | tile_dim2=0
    g1[5] = (int)stride0;                                    // dim0_stride[31:0]
    g1[6] = 0;                                               // stride0 hi | stride1 lo
    g1[7] = 0;
    i32x4 z4 = {0, 0, 0, 0};
    i32x8 z8 = {0, 0, 0, 0, 0, 0, 0, 0};
    __builtin_amdgcn_tensor_load_to_lds(g0, g1, z4, z4, z8, 0);
}

__device__ __forceinline__ unsigned lds_off_of(const void* p) {
    // LDS aperture: addr[31:0] is the LDS byte offset (ISA 10.2 aperture mapping)
    return (unsigned)(unsigned long long)p;
}

// order-preserving float -> u32 key
__device__ __forceinline__ unsigned int fkey(float f) {
    unsigned int u = __float_as_uint(f);
    return (u & 0x80000000u) ? ~u : (u | 0x80000000u);
}

__device__ __forceinline__ void top6_insert(unsigned long long k, unsigned long long t[6]) {
    if (k <= t[5]) return;
    t[5] = k;
#pragma unroll
    for (int a = 5; a > 0; --a) {
        if (t[a] > t[a - 1]) { unsigned long long tmp = t[a]; t[a] = t[a - 1]; t[a - 1] = tmp; }
    }
}

// ---------------------------------------------------------------- conversions
__global__ void k_convert_f16(const float* __restrict__ in, _Float16* __restrict__ out, int n) {
    for (int i = blockIdx.x * 256 + threadIdx.x; i < n; i += gridDim.x * 256)
        out[i] = (_Float16)in[i];
}

__global__ void k_transpose_f16(const float* __restrict__ in, _Float16* __restrict__ out,
                                int rows_in, int cols_in) {
    int n = rows_in * cols_in;
    for (int i = blockIdx.x * 256 + threadIdx.x; i < n; i += gridDim.x * 256) {
        int c = i / rows_in;
        int r = i % rows_in;
        out[(long)c * rows_in + r] = (_Float16)in[(long)r * cols_in + c];
    }
}

// ---------------------------------------------------------------- encoder GEMMs
__global__ void k_gemm_lrelu(const _Float16* __restrict__ A, const _Float16* __restrict__ Bt,
                             const float* __restrict__ bias, _Float16* __restrict__ H1) {
    int gid = blockIdx.x * 8 + (threadIdx.x >> 5);
    int lane = threadIdx.x & 31;
    int row0 = (gid >> 5) * 16, col0 = (gid & 31) * 16;
    int n = lane & 15, half = lane >> 4;
    v8f c = gemm_tile(A, row0 + n, MMD, Bt, col0 + n, MMD, MMD, half);
#pragma unroll
    for (int r = 0; r < 8; ++r) {
        int row = row0 + r + 8 * half;
        int col = col0 + n;
        float x = c[r] + bias[col];
        x = x > 0.f ? x : 0.01f * x;
        H1[(long)row * DD + col] = (_Float16)x;
    }
}

__global__ void k_gemm_bias(const _Float16* __restrict__ A, const _Float16* __restrict__ Bt,
                            const float* __restrict__ bias, float* __restrict__ H2) {
    int gid = blockIdx.x * 8 + (threadIdx.x >> 5);
    int lane = threadIdx.x & 31;
    int row0 = (gid >> 5) * 16, col0 = (gid & 31) * 16;
    int n = lane & 15, half = lane >> 4;
    v8f c = gemm_tile(A, row0 + n, DD, Bt, col0 + n, DD, DD, half);
#pragma unroll
    for (int r = 0; r < 8; ++r)
        H2[(long)(row0 + r + 8 * half) * DD + col0 + n] = c[r] + bias[col0 + n];
}

// ---------------------------------------------------------------- BatchNorm stats
__global__ void k_colstats(const float* __restrict__ H2, const float* __restrict__ gamma,
                           const float* __restrict__ beta, float* __restrict__ scale,
                           float* __restrict__ shift) {
    __shared__ float rs[256], rq[256];
    int c = blockIdx.x, t = threadIdx.x;
    float s = 0.f, q = 0.f;
    for (int r = t; r < N_MM; r += 256) {
        float v = H2[(long)r * DD + c];
        s += v; q += v * v;
    }
    rs[t] = s; rq[t] = q; __syncthreads();
    for (int o = 128; o > 0; o >>= 1) {
        if (t < o) { rs[t] += rs[t + o]; rq[t] += rq[t + o]; }
        __syncthreads();
    }
    if (t == 0) {
        float mu  = rs[0] * (1.0f / N_MM);
        float var = rq[0] * (1.0f / N_MM) - mu * mu;
        float scl = gamma[c] * rsqrtf(var + BN_EPS);
        scale[c] = scl;
        shift[c] = beta[c] - mu * scl;
    }
}

__global__ void k_bn_rownorm(const float* __restrict__ H2, const float* __restrict__ scale,
                             const float* __restrict__ shift, float* __restrict__ enc,
                             _Float16* __restrict__ MN) {
    __shared__ float red[256];
    int r = blockIdx.x, t = threadIdx.x;
    int c0 = t, c1 = t + 256;
    float e0 = H2[(long)r * DD + c0] * scale[c0] + shift[c0];
    float e1 = H2[(long)r * DD + c1] * scale[c1] + shift[c1];
    enc[(long)r * DD + c0] = e0;
    enc[(long)r * DD + c1] = e1;
    red[t] = e0 * e0 + e1 * e1; __syncthreads();
    for (int o = 128; o > 0; o >>= 1) {
        if (t < o) red[t] += red[t + o];
        __syncthreads();
    }
    float inv = 1.0f / (sqrtf(red[0]) + 1e-12f);
    MN[(long)r * DD + c0] = (_Float16)(e0 * inv);
    MN[(long)r * DD + c1] = (_Float16)(e1 * inv);
}

__global__ void k_strownorm_init(const float* __restrict__ ST, _Float16* __restrict__ SN,
                                 float* __restrict__ rowsum, unsigned long long* __restrict__ amax,
                                 float* __restrict__ loss_out) {
    __shared__ float red[256];
    int r = blockIdx.x, t = threadIdx.x;
    int c0 = t, c1 = t + 256;
    float e0 = ST[(long)r * DD + c0];
    float e1 = ST[(long)r * DD + c1];
    red[t] = e0 * e0 + e1 * e1; __syncthreads();
    for (int o = 128; o > 0; o >>= 1) {
        if (t < o) red[t] += red[t + o];
        __syncthreads();
    }
    float inv = 1.0f / (sqrtf(red[0]) + 1e-12f);
    SN[(long)r * DD + c0] = (_Float16)(e0 * inv);
    SN[(long)r * DD + c1] = (_Float16)(e1 * inv);
    if (t == 0) { rowsum[r] = 0.f; amax[r] = 0ull; }
    if (r == 0 && t == 0) loss_out[0] = 0.f;
}

// ---------------------------------------------------------------- similarity pass
// grid (64 j-chunks, 512 row-tiles); block = 8 waves.
// A chunk (16x32) + B panel chunk (128x32) staged in LDS by the TDM,
// double-buffered on TENSORcnt; WMMA reads both operands from LDS.
__global__ void k_sim(const _Float16* __restrict__ SN, const _Float16* __restrict__ MN,
                      float* __restrict__ rowsum, unsigned long long* __restrict__ amax) {
    __shared__ _Float16 ldsA[2][16 * 32];
    __shared__ _Float16 ldsB[2][128 * 32];
    __shared__ float stile[8][16][16];
    __shared__ float red_s[16][16];
    __shared__ unsigned long long red_k[16][16];

    int w = threadIdx.x >> 5, lane = threadIdx.x & 31;
    int row0  = blockIdx.y * 16;
    int jbase = blockIdx.x * 128;
    int n = lane & 15, half = lane >> 4;

    if (threadIdx.x < 32) {   // wave 0 drives the TDM pipeline
        tdm_load_2d(lds_off_of(ldsA[0]), SN + (long)row0 * DD, DD, N_ST, 32, 16, DD);
        tdm_load_2d(lds_off_of(ldsB[0]), MN + (long)jbase * DD, DD, N_MM, 32, 128, DD);
    }

    v8f c = {};
    for (int kc = 0; kc < 16; ++kc) {
        int buf = kc & 1;
        if (threadIdx.x < 32) {
            if (kc < 15) {
                int k0 = (kc + 1) * 32;
                tdm_load_2d(lds_off_of(ldsA[buf ^ 1]), SN + (long)row0 * DD + k0,
                            DD, N_ST, 32, 16, DD);
                tdm_load_2d(lds_off_of(ldsB[buf ^ 1]), MN + (long)jbase * DD + k0,
                            DD, N_MM, 32, 128, DD);
                __builtin_amdgcn_s_wait_tensorcnt(2);  // chunk kc resident
            } else {
                __builtin_amdgcn_s_wait_tensorcnt(0);
            }
        }
        __syncthreads();
        HFrag fa;
        const _Float16* pa = &ldsA[buf][n * 32 + half * 8];
        fa.q[0] = *(const uint4*)pa;
        fa.q[1] = *(const uint4*)(pa + 16);
        HFrag fb;
        const _Float16* pb = &ldsB[buf][(w * 16 + n) * 32 + half * 8];
        fb.q[0] = *(const uint4*)pb;
        fb.q[1] = *(const uint4*)(pb + 16);
        c = wmma_f16(fa.v, fb.v, c);
        __syncthreads();
    }

#pragma unroll
    for (int r = 0; r < 8; ++r) stile[w][r + 8 * half][n] = c[r];
    __syncthreads();

    int row = threadIdx.x & 15, seg = threadIdx.x >> 4;
    int wv = seg >> 1, cb = (seg & 1) * 8;
    float s = 0.f; unsigned long long bk = 0ull;
#pragma unroll
    for (int cc = 0; cc < 8; ++cc) {
        float v = stile[wv][row][cb + cc];
        int j = jbase + wv * 16 + cb + cc;
        s += expf(2.0f * v);
        unsigned long long key =
            ((unsigned long long)fkey(v) << 32) | (unsigned int)(0x7FFFFFFF - j);
        if (key > bk) bk = key;
    }
    red_s[row][seg] = s;
    red_k[row][seg] = bk;
    __syncthreads();
    if (threadIdx.x < 16) {
        int rr = threadIdx.x;
        float tot = 0.f; unsigned long long best = 0ull;
#pragma unroll
        for (int g = 0; g < 16; ++g) {
            tot += red_s[rr][g];
            if (red_k[rr][g] > best) best = red_k[rr][g];
        }
        atomicAdd(&rowsum[row0 + rr], tot);
        atomicMax(&amax[row0 + rr], best);
    }
}

// ---------------------------------------------------------------- top-6 pass
// gathered A rows staged once in LDS; B streamed from global (L2-resident)
__global__ void k_topk(const _Float16* __restrict__ MN,
                       const unsigned long long* __restrict__ amax,
                       int* __restrict__ topkidx) {
    __shared__ int ridx[16];
    __shared__ _Float16 ldsA[16 * DD];     // 16 KB gathered A panel
    __shared__ float stile[8][16][16];
    __shared__ unsigned long long topbuf[8][16][TOPKN];
    int row0 = blockIdx.x * 16;
    int w = threadIdx.x >> 5, lane = threadIdx.x & 31;
    if (threadIdx.x < 16) {
        unsigned long long key = amax[row0 + threadIdx.x];
        ridx[threadIdx.x] = 0x7FFFFFFF - (int)(key & 0xFFFFFFFFull);
    }
    __syncthreads();
    {   // cooperative gather: 64B per thread
        int rr = threadIdx.x >> 4, ck = threadIdx.x & 15;
        const uint4* src = (const uint4*)(MN + (long)ridx[rr] * DD + ck * 32);
        uint4* dst = (uint4*)(&ldsA[rr * DD + ck * 32]);
#pragma unroll
        for (int u = 0; u < 4; ++u) dst[u] = src[u];
    }
    __syncthreads();

    int m = lane & 15, half = lane >> 4;
    unsigned long long t6[TOPKN] = {0, 0, 0, 0, 0, 0};

    for (int jc = 0; jc < 64; ++jc) {
        int col0 = jc * 128 + w * 16;
        if (jc < 63)  // hint next B panel toward the caches
            __builtin_prefetch(MN + (long)(col0 + 128 + m) * DD, 0, 0);
        const _Float16* bp = MN + (long)(col0 + m) * DD;
        v8f c = {};
        for (int k0 = 0; k0 < DD; k0 += 32) {
            HFrag fa;
            const _Float16* pa = &ldsA[m * DD + k0 + half * 8];
            fa.q[0] = *(const uint4*)pa;
            fa.q[1] = *(const uint4*)(pa + 16);
            c = wmma_f16(fa.v, load_b_frag(bp, k0, half), c);
        }
#pragma unroll
        for (int r = 0; r < 8; ++r) stile[w][r + 8 * half][m] = c[r];
        __syncthreads();
        if (lane < 16) {
#pragma unroll
            for (int cc = 0; cc < 16; ++cc) {
                float v = stile[w][lane][cc];
                unsigned long long key =
                    ((unsigned long long)fkey(v) << 32) |
                    (unsigned int)(0x7FFFFFFF - (col0 + cc));
                top6_insert(key, t6);
            }
        }
        __syncthreads();
    }
    if (lane < 16) {
#pragma unroll
        for (int p = 0; p < TOPKN; ++p) topbuf[w][lane][p] = t6[p];
    }
    __syncthreads();
    if (threadIdx.x < 16) {
        int rr = threadIdx.x;
        for (int p = 0; p < TOPKN; ++p) {
            unsigned long long best = 0ull; int bw = 0, be = 0;
            for (int ww = 0; ww < 8; ++ww)
                for (int e = 0; e < TOPKN; ++e) {
                    unsigned long long k = topbuf[ww][rr][e];
                    if (k > best) { best = k; bw = ww; be = e; }
                }
            topbuf[bw][rr][be] = 0ull;
            topkidx[(row0 + rr) * TOPKN + p] = 0x7FFFFFFF - (int)(best & 0xFFFFFFFFull);
        }
    }
}

// ---------------------------------------------------------------- loss
__global__ void k_loss(const _Float16* __restrict__ SN, const _Float16* __restrict__ MN,
                       const int* __restrict__ topkidx, const float* __restrict__ rowsum,
                       float* __restrict__ loss_out) {
    __shared__ float red[256];
    __shared__ float dots[TOPKN];
    int row = blockIdx.x, t = threadIdx.x;
    int idx[TOPKN];
#pragma unroll
    for (int p = 0; p < TOPKN; ++p) idx[p] = topkidx[row * TOPKN + p];
    float acc[TOPKN] = {0, 0, 0, 0, 0, 0};
    for (int c = t; c < DD; c += 256) {
        float sv = (float)SN[(long)row * DD + c];
#pragma unroll
        for (int p = 0; p < TOPKN; ++p)
            acc[p] += sv * (float)MN[(long)idx[p] * DD + c];
    }
    for (int p = 0; p < TOPKN; ++p) {
        red[t] = acc[p]; __syncthreads();
        for (int o = 128; o > 0; o >>= 1) {
            if (t < o) red[t] += red[t + o];
            __syncthreads();
        }
        if (t == 0) dots[p] = red[0];
        __syncthreads();
    }
    if (t == 0) {
        float ps = 0.f;
#pragma unroll
        for (int p = 0; p < TOPKN; ++p) ps += expf(2.0f * dots[p]);
        float li = logf(rowsum[row]) - logf(ps);
        atomicAdd(loss_out, li * (1.0f / N_ST));
    }
}

// ---------------------------------------------------------------- launch
extern "C" void kernel_launch(void* const* d_in, const int* in_sizes, int n_in,
                              void* d_out, int out_size, void* d_ws, size_t ws_size,
                              hipStream_t stream) {
    const float* st    = (const float*)d_in[0];
    const float* mm    = (const float*)d_in[1];
    const float* W1    = (const float*)d_in[2];
    const float* b1    = (const float*)d_in[3];
    const float* W2    = (const float*)d_in[4];
    const float* b2    = (const float*)d_in[5];
    const float* gamma = (const float*)d_in[6];
    const float* beta  = (const float*)d_in[7];
    float* out = (float*)d_out;

    char* ws = (char*)d_ws;
    size_t off = 0;
    auto alloc = [&](size_t bytes) {
        void* p = ws + off;
        off = (off + bytes + 255) & ~(size_t)255;
        return p;
    };
    _Float16* mm16  = (_Float16*)alloc((size_t)N_MM * MMD * 2);
    _Float16* W1t   = (_Float16*)alloc((size_t)DD * MMD * 2);
    _Float16* W2t   = (_Float16*)alloc((size_t)DD * DD * 2);
    _Float16* h1    = (_Float16*)alloc((size_t)N_MM * DD * 2);
    float*    h2    = (float*)   alloc((size_t)N_MM * DD * 4);
    float*    scale = (float*)   alloc((size_t)DD * 4);
    float*    shift = (float*)   alloc((size_t)DD * 4);
    _Float16* mn16  = (_Float16*)alloc((size_t)N_MM * DD * 2);
    _Float16* sn16  = (_Float16*)alloc((size_t)N_ST * DD * 2);
    float*    rowsum = (float*)  alloc((size_t)N_ST * 4);
    unsigned long long* amax = (unsigned long long*)alloc((size_t)N_ST * 8);
    int*      topkidx = (int*)   alloc((size_t)N_ST * TOPKN * 4);

    float* loss_slot = out + (size_t)N_MM * DD;

    k_convert_f16  <<<4096, 256, 0, stream>>>(mm, mm16, N_MM * MMD);
    k_transpose_f16<<<1536, 256, 0, stream>>>(W1, W1t, MMD, DD);
    k_transpose_f16<<<1024, 256, 0, stream>>>(W2, W2t, DD, DD);

    k_gemm_lrelu<<<2048, 256, 0, stream>>>(mm16, W1t, b1, h1);
    k_gemm_bias <<<2048, 256, 0, stream>>>(h1, W2t, b2, h2);

    k_colstats  <<<DD, 256, 0, stream>>>(h2, gamma, beta, scale, shift);
    k_bn_rownorm<<<N_MM, 256, 0, stream>>>(h2, scale, shift, out, mn16);
    k_strownorm_init<<<N_ST, 256, 0, stream>>>(st, sn16, rowsum, amax, loss_slot);

    k_sim <<<dim3(64, 512), 256, 0, stream>>>(sn16, mn16, rowsum, amax);
    k_topk<<<512, 256, 0, stream>>>(mn16, amax, topkidx);
    k_loss<<<N_ST, 256, 0, stream>>>(sn16, mn16, topkidx, rowsum, loss_slot);
}
